// LearnableMultiheadSelfAttention_46385646797428
// MI455X (gfx1250) — matrix operand order repaired and checked by reading, not compiled, not run on hardware
//
#include <hip/hip_runtime.h>

typedef __bf16 bf16;
typedef __attribute__((ext_vector_type(16))) __bf16 v16bf;
typedef __attribute__((ext_vector_type(8)))  __bf16 v8bf;
typedef __attribute__((ext_vector_type(4)))  __bf16 v4bf;
typedef __attribute__((ext_vector_type(8)))  float  v8f;

#define S_LEN 1024
#define M_LEN 1024
#define T_LEN 2048
#define BATCH 4
#define HEADS 16
#define DHEAD 64
#define DMODEL 1024
#define NEGBIG (-1.0e30f)

#if defined(__has_builtin)
# if __has_builtin(__builtin_amdgcn_tensor_load_to_lds) && __has_builtin(__builtin_amdgcn_s_wait_tensorcnt)
#  define HAVE_TDM 1
# endif
#endif
#ifndef HAVE_TDM
# define HAVE_TDM 0
#endif

union FragU { v16bf v; struct { v8bf lo, hi; } h; };

// A-matrix 16x32 bf16 fragment (ISA 7.12.2 layout): lanes 0-15 row M=lane,
// elems 0..7 = K0..7, elems 8..15 = K16..23; lanes 16-31 get K8..15 / K24..31.
__device__ inline v16bf load_frag_a(const bf16* base, int ld) {
    int lane = threadIdx.x & 31;
    int m = lane & 15, half = lane >> 4;
    const bf16* p = base + m * ld + half * 8;
    FragU f;
    f.h.lo = *(const v8bf*)(p);
    f.h.hi = *(const v8bf*)(p + 16);
    return f.v;
}

// B-matrix 32x16 bf16 fragment from an [N][K] (column-major-of-B) LDS tile:
// lanes 0-15: N=lane, K=0..7,16..23; lanes 16-31: N=lane-16, K=8..15,24..31.
__device__ inline v16bf load_frag_b(const bf16* base, int ld) {
    int lane = threadIdx.x & 31;
    int n = lane & 15, half = lane >> 4;
    const bf16* p = base + n * ld + half * 8;
    FragU f;
    f.h.lo = *(const v8bf*)(p);
    f.h.hi = *(const v8bf*)(p + 16);
    return f.v;
}

__device__ inline v8f wmma_bf16(v16bf a, v16bf b, v8f c) {
    return __builtin_amdgcn_wmma_f32_16x16x32_bf16(false, a, false, b, (short)0, c, false, false);
}

// ---------------------------------------------------------------------------
// Tensor Data Mover: 2-D tile load Global -> LDS via D# descriptor
// (CDNA5 ISA). data_size fixed to 2 bytes (bf16).
// pad_interval_code/pad_amount_code per §8.4 (4 => 32 DWORDs, 3 => 4 DWORDs).
// ---------------------------------------------------------------------------
__device__ inline void tdm_load_2d_bf16(const void* gsrc, void* ldst,
                                        unsigned tile_d0, unsigned tile_d1,
                                        unsigned long long stride0_elems,
                                        unsigned tensor_d0, unsigned tensor_d1,
                                        unsigned pad_interval_code,
                                        unsigned pad_amount_code, bool pad_en) {
#if HAVE_TDM
    typedef unsigned tdm_v4u __attribute__((ext_vector_type(4)));
    typedef int      tdm_v8i __attribute__((ext_vector_type(8)));
    typedef int      tdm_v4i __attribute__((ext_vector_type(4)));
    unsigned long long ga = (unsigned long long)(size_t)gsrc;
    unsigned lds = (unsigned)(size_t)ldst;
    tdm_v4u g0;
    g0[0] = 1u;                                   // count=1, user descriptor
    g0[1] = lds;                                  // lds_addr [63:32]
    g0[2] = (unsigned)ga;                         // global_addr [95:64]
    g0[3] = (unsigned)((ga >> 32) & 0x01FFFFFFu)  // global_addr [120:96]
          | (2u << 30);                           // type=2 ("image")
    tdm_v8i g1;
    g1[0] = (int)((1u << 16)                                   // data_size=2B
                | (pad_en ? (1u << 20) : 0u)                   // pad_enable
                | (pad_interval_code << 22)
                | (pad_amount_code << 25));
    g1[1] = (int)((tensor_d0 & 0xFFFFu) << 16);                // td0[15:0]
    g1[2] = (int)((tensor_d0 >> 16) | ((tensor_d1 & 0xFFFFu) << 16));
    g1[3] = (int)((tensor_d1 >> 16) | (tile_d0 << 16));        // tile_dim0
    g1[4] = (int)tile_d1;                                      // tile_dim1 (dim2=0)
    g1[5] = (int)(unsigned)(stride0_elems & 0xFFFFFFFFull);    // dim0 stride
    g1[6] = (int)(unsigned)((stride0_elems >> 32) & 0xFFFFull);
    g1[7] = 0;
    tdm_v4i z4 = (tdm_v4i){0, 0, 0, 0};
#if __clang_major__ >= 23
    tdm_v8i z8 = (tdm_v8i){0, 0, 0, 0, 0, 0, 0, 0};
    __builtin_amdgcn_tensor_load_to_lds(g0, g1, z4, z4, z8, 0);
#else
    __builtin_amdgcn_tensor_load_to_lds(g0, g1, z4, z4, 0);
#endif
#else
    (void)gsrc; (void)ldst; (void)tile_d0; (void)tile_d1; (void)stride0_elems;
    (void)tensor_d0; (void)tensor_d1; (void)pad_interval_code;
    (void)pad_amount_code; (void)pad_en;
#endif
}

// ---------------------------------------------------------------------------
// Generic GEMM: C[M,N] = A[M,K] (f32, row-major) * B[K,N] (f32, row-major).
// 128x128 block tile, 8 waves (2 col x 4 row), bf16 WMMA with f32 accum.
// ---------------------------------------------------------------------------
template<bool OUT_BF16>
__global__ __launch_bounds__(256)
void gemm_tn(const float* __restrict__ A, const float* __restrict__ B,
             void* __restrict__ Cout, int Mdim, int Ndim, int Kdim) {
    __shared__ bf16 As[128 * 40];   // [m][k], ld 40
    __shared__ bf16 Bs[128 * 40];   // [n][k] (transposed during staging), ld 40
    const int ldT = 40;
    const int tid = threadIdx.x;
    const int wave = tid >> 5, lane = tid & 31;
    const int m0 = blockIdx.x * 128;
    const int n0 = blockIdx.y * 128;
    const int wm = (wave & 3) * 32;   // wave row offset
    const int wn = (wave >> 2) * 64;  // wave col offset

    v8f acc[2][4];
#pragma unroll
    for (int i = 0; i < 2; ++i)
#pragma unroll
        for (int j = 0; j < 4; ++j) acc[i][j] = (v8f){};

    for (int k0 = 0; k0 < Kdim; k0 += 32) {
        // stage A 128x32 (f32 -> bf16)
#pragma unroll
        for (int it = 0; it < 4; ++it) {
            int idx = it * 256 + tid;
            int r = idx >> 3, c4 = idx & 7;
            float4 f = *(const float4*)(A + (size_t)(m0 + r) * Kdim + k0 + c4 * 4);
            v4bf o = { (bf16)f.x, (bf16)f.y, (bf16)f.z, (bf16)f.w };
            *(v4bf*)(&As[r * ldT + c4 * 4]) = o;
        }
        // stage B 32x128 transposed into [n][k]
#pragma unroll
        for (int it = 0; it < 4; ++it) {
            int idx = it * 256 + tid;
            int kk = idx >> 5, n4 = (idx & 31) * 4;
            float4 f = *(const float4*)(B + (size_t)(k0 + kk) * Ndim + n0 + n4);
            Bs[(n4 + 0) * ldT + kk] = (bf16)f.x;
            Bs[(n4 + 1) * ldT + kk] = (bf16)f.y;
            Bs[(n4 + 2) * ldT + kk] = (bf16)f.z;
            Bs[(n4 + 3) * ldT + kk] = (bf16)f.w;
        }
        if (k0 + 32 < Kdim)  // hint next A tile into L2 (global_prefetch_b8)
            __builtin_prefetch(A + (size_t)(m0 + ((tid * 4) >> 3)) * Kdim + k0 + 32, 0, 1);
        __syncthreads();

        v16bf af[2], bfr[4];
#pragma unroll
        for (int i = 0; i < 2; ++i) af[i] = load_frag_a(As + (wm + i * 16) * ldT, ldT);
#pragma unroll
        for (int j = 0; j < 4; ++j) bfr[j] = load_frag_b(Bs + (wn + j * 16) * ldT, ldT);
#pragma unroll
        for (int i = 0; i < 2; ++i)
#pragma unroll
            for (int j = 0; j < 4; ++j) acc[i][j] = wmma_bf16(af[i], bfr[j], acc[i][j]);
        __syncthreads();
    }

    const int nloc = lane & 15, rhalf = (lane >> 4) * 8;
#pragma unroll
    for (int i = 0; i < 2; ++i)
#pragma unroll
        for (int j = 0; j < 4; ++j)
#pragma unroll
            for (int g = 0; g < 8; ++g) {
                size_t row = m0 + wm + i * 16 + rhalf + g;
                size_t col = n0 + wn + j * 16 + nloc;
                float v = acc[i][j][g];
                if (OUT_BF16) ((bf16*)Cout)[row * Ndim + col] = (bf16)v;
                else          ((float*)Cout)[row * Ndim + col] = v;
            }
}

// ---------------------------------------------------------------------------
// Fused Transformer-XL attention (flash-style, online softmax).
// grid = (S/64, B, H), 128 threads = 4 waves, each wave owns 16 query rows.
// K tile and r window staged by the Tensor Data Mover (wave 0 issues the D#,
// waits TENSORcnt, block barrier publishes). BD relative shift: score(i,j)
// uses r[j - i + S - 1]; computed via WMMA over an 80-wide jj-window per wave,
// staged through LDS, gathered diagonally.
// ---------------------------------------------------------------------------
__global__ __launch_bounds__(128)
void attn_xl_kernel(const bf16* __restrict__ qkv,   // [T*B, 3*H*DH] bf16
                    const bf16* __restrict__ rmat,  // [T, H*DH] bf16
                    const float* __restrict__ bias_u,
                    const float* __restrict__ bias_v,
                    float* __restrict__ attn_vec) { // [S*B, H*DH] f32
    __shared__ bf16 qu[64 * 72];     // (q+u) bf16 [i][d]
    __shared__ bf16 qv[64 * 72];     // (q+v) bf16 [i][d]
    __shared__ bf16 ks[64 * 72];     // K tile  [j][d]
    __shared__ bf16 vs[64 * 72];     // V tile transposed [d][j]
    __shared__ bf16 rs[128 * 72];    // r window [jj][d]
    __shared__ bf16 ps[4][16 * 72];  // probs bf16 per wave [i][j]
    __shared__ float bd[4][16 * 80]; // BD window per wave [i][t]

    const int i0 = blockIdx.x * 64;
    const int b  = blockIdx.y;
    const int h  = blockIdx.z;
    const int tid = threadIdx.x;
    const int wv = tid >> 5, lane = tid & 31;
    const int nloc = lane & 15, half = lane >> 4;
    const size_t qld = 3 * HEADS * DHEAD;

    // stage q+u, q+v (bf16 q promoted to f32, bias added, back to bf16)
    for (int idx = tid; idx < 64 * 32; idx += 128) {
        int row = idx >> 5;
        int dp  = (idx & 31) * 2;
        const bf16* qp = qkv + ((size_t)(M_LEN + i0 + row) * BATCH + b) * qld + h * DHEAD + dp;
        float q0 = (float)qp[0], q1 = (float)qp[1];
        qu[row * 72 + dp]     = (bf16)(q0 + bias_u[h * DHEAD + dp]);
        qu[row * 72 + dp + 1] = (bf16)(q1 + bias_u[h * DHEAD + dp + 1]);
        qv[row * 72 + dp]     = (bf16)(q0 + bias_v[h * DHEAD + dp]);
        qv[row * 72 + dp + 1] = (bf16)(q1 + bias_v[h * DHEAD + dp + 1]);
    }

    float mrow[8], lrow[8];
    v8f o[4];
#pragma unroll
    for (int g = 0; g < 8; ++g) { mrow[g] = NEGBIG; lrow[g] = 0.f; }
#pragma unroll
    for (int d = 0; d < 4; ++d) o[d] = (v8f){};

    const int jt_max = (i0 + 63 + M_LEN) >> 6;  // last tile with any unmasked col
    for (int jt = 0; jt <= jt_max; ++jt) {
        const int j0 = jt * 64;
        const int rstart = j0 - i0 + S_LEN - 64;   // >= 0 for all live tiles
        __syncthreads();
#if HAVE_TDM
        if (wv == 0) {
            // K tile: 64 rows (stride B*qld elems between j's) x 64 bf16,
            // padded to 72-element LDS rows (32 DW data + 4 DW pad).
            const bf16* ksrc = qkv + ((size_t)j0 * BATCH + b) * qld
                               + HEADS * DHEAD + h * DHEAD;
            tdm_load_2d_bf16(ksrc, ks, 64, 64,
                             (unsigned long long)(BATCH * 3 * HEADS * DHEAD),
                             64, 64, 4, 3, true);
            // r window: 128 rows x 64 bf16; rows past T read zero via OOB.
            const bf16* rsrc = rmat + (size_t)rstart * (HEADS * DHEAD) + h * DHEAD;
            unsigned rows_valid = (unsigned)(T_LEN - rstart);
            tdm_load_2d_bf16(rsrc, rs, 64, 128,
                             (unsigned long long)(HEADS * DHEAD),
                             64, rows_valid, 4, 3, true);
            __builtin_amdgcn_s_wait_tensorcnt(0);
        }
#else
        // stage K tile [j][d] (uint4 = 8 bf16)
#pragma unroll
        for (int it = 0; it < 4; ++it) {
            int idx = it * 128 + tid;
            int row = idx >> 3, c = (idx & 7) * 8;
            *(uint4*)&ks[row * 72 + c] =
                *(const uint4*)(qkv + ((size_t)(j0 + row) * BATCH + b) * qld + HEADS * DHEAD + h * DHEAD + c);
        }
        // stage r window: rows rstart..rstart+127 (clamped; OOB rows are masked)
#pragma unroll
        for (int it = 0; it < 8; ++it) {
            int idx = it * 128 + tid;
            int row = idx >> 3, c = (idx & 7) * 8;
            int rr = rstart + row;
            rr = rr < 0 ? 0 : (rr > T_LEN - 1 ? T_LEN - 1 : rr);
            *(uint4*)&rs[row * 72 + c] =
                *(const uint4*)(rmat + (size_t)rr * (HEADS * DHEAD) + h * DHEAD + c);
        }
#endif
        // stage V transposed [d][j] (needs transpose+layout change: manual)
        for (int idx = tid; idx < 64 * 32; idx += 128) {
            int row = idx >> 5;
            int dp  = (idx & 31) * 2;
            const bf16* vp = qkv + ((size_t)(j0 + row) * BATCH + b) * qld + 2 * HEADS * DHEAD + h * DHEAD + dp;
            vs[dp * 72 + row]       = vp[0];
            vs[(dp + 1) * 72 + row] = vp[1];
        }
        __syncthreads();

        // --- AC = (q+u) K^T : 16x64 per wave ---
        v16bf aq[2], av[2];
#pragma unroll
        for (int kc = 0; kc < 2; ++kc) {
            aq[kc] = load_frag_a(qu + (wv * 16) * 72 + kc * 32, 72);
            av[kc] = load_frag_a(qv + (wv * 16) * 72 + kc * 32, 72);
        }
        v8f sAC[4];
#pragma unroll
        for (int ns = 0; ns < 4; ++ns) {
            v8f a = (v8f){};
#pragma unroll
            for (int kc = 0; kc < 2; ++kc)
                a = wmma_bf16(aq[kc], load_frag_b(ks + (ns * 16) * 72 + kc * 32, 72), a);
            sAC[ns] = a;
        }
        // --- BD window (5 tiles of 16 = 80 jj values) -> LDS ---
        const int woff = 48 - wv * 16;  // wave window base inside rs
#pragma unroll
        for (int tt = 0; tt < 5; ++tt) {
            v8f a = (v8f){};
#pragma unroll
            for (int kc = 0; kc < 2; ++kc)
                a = wmma_bf16(av[kc], load_frag_b(rs + (woff + tt * 16) * 72 + kc * 32, 72), a);
#pragma unroll
            for (int g = 0; g < 8; ++g)
                bd[wv][(g + half * 8) * 80 + tt * 16 + nloc] = a[g];
        }

        // --- assemble scores (diagonal-shift gather) + online softmax ---
        float sv[4][8], rmax[8];
#pragma unroll
        for (int g = 0; g < 8; ++g) rmax[g] = NEGBIG;
#pragma unroll
        for (int ns = 0; ns < 4; ++ns)
#pragma unroll
            for (int g = 0; g < 8; ++g) {
                int rloc = g + half * 8;
                int jj = j0 + ns * 16 + nloc;
                int ii = i0 + wv * 16 + rloc;
                float s = NEGBIG;
                if (jj <= ii + M_LEN) {
                    float bdv = bd[wv][rloc * 80 + ns * 16 + nloc + 15 - rloc];
                    s = (sAC[ns][g] + bdv) * 0.125f;  // 1/sqrt(64)
                }
                sv[ns][g] = s;
                rmax[g] = fmaxf(rmax[g], s);
            }
#pragma unroll
        for (int m = 1; m < 16; m <<= 1)
#pragma unroll
            for (int g = 0; g < 8; ++g) rmax[g] = fmaxf(rmax[g], __shfl_xor(rmax[g], m));

        float alpha[8], rsum[8];
#pragma unroll
        for (int g = 0; g < 8; ++g) {
            float mo = mrow[g];
            float mn = fmaxf(mo, rmax[g]);
            mrow[g] = mn;
            alpha[g] = __expf(mo - mn);   // 0 at first tile (mo=-1e30)
            rsum[g] = 0.f;
        }
#pragma unroll
        for (int ns = 0; ns < 4; ++ns)
#pragma unroll
            for (int g = 0; g < 8; ++g) {
                float p = __expf(sv[ns][g] - mrow[g]);  // masked -> underflow 0
                sv[ns][g] = p;
                rsum[g] += p;
            }
#pragma unroll
        for (int m = 1; m < 16; m <<= 1)
#pragma unroll
            for (int g = 0; g < 8; ++g) rsum[g] += __shfl_xor(rsum[g], m);
#pragma unroll
        for (int g = 0; g < 8; ++g) lrow[g] = lrow[g] * alpha[g] + rsum[g];
#pragma unroll
        for (int d = 0; d < 4; ++d)
#pragma unroll
            for (int g = 0; g < 8; ++g) o[d][g] *= alpha[g];

        // probs -> bf16 LDS, then O += P V via WMMA
#pragma unroll
        for (int ns = 0; ns < 4; ++ns)
#pragma unroll
            for (int g = 0; g < 8; ++g)
                ps[wv][(g + half * 8) * 72 + ns * 16 + nloc] = (bf16)sv[ns][g];

        v16bf ap[2];
#pragma unroll
        for (int kc = 0; kc < 2; ++kc) ap[kc] = load_frag_a(&ps[wv][0] + kc * 32, 72);
#pragma unroll
        for (int d = 0; d < 4; ++d)
#pragma unroll
            for (int kc = 0; kc < 2; ++kc)
                o[d] = wmma_bf16(ap[kc], load_frag_b(vs + (d * 16) * 72 + kc * 32, 72), o[d]);
    }

    // epilogue: normalize rows, write attn_vec
#pragma unroll
    for (int g = 0; g < 8; ++g) lrow[g] = 1.0f / lrow[g];
#pragma unroll
    for (int d = 0; d < 4; ++d)
#pragma unroll
        for (int g = 0; g < 8; ++g) {
            size_t row = (size_t)(i0 + wv * 16 + g + half * 8) * BATCH + b;
            attn_vec[row * (HEADS * DHEAD) + h * DHEAD + d * 16 + nloc] = o[d][g] * lrow[g];
        }
}

// ---------------------------------------------------------------------------
// Residual + LayerNorm: out = LN(x + attn_out) * gamma + beta
// ---------------------------------------------------------------------------
__global__ __launch_bounds__(256)
void ln_kernel(const float* __restrict__ x, const float* __restrict__ ao,
               const float* __restrict__ gamma, const float* __restrict__ beta,
               float* __restrict__ out) {
    const int row = blockIdx.x;
    const int tid = threadIdx.x;
    const float* xr = x + (size_t)row * DMODEL;
    const float* ar = ao + (size_t)row * DMODEL;
    __shared__ float red[2][8];
    float yv[4], s = 0.f, s2 = 0.f;
#pragma unroll
    for (int q = 0; q < 4; ++q) {
        int c = tid + q * 256;
        float y = xr[c] + ar[c];
        yv[q] = y; s += y; s2 += y * y;
    }
#pragma unroll
    for (int m = 1; m < 32; m <<= 1) { s += __shfl_xor(s, m); s2 += __shfl_xor(s2, m); }
    if ((tid & 31) == 0) { red[0][tid >> 5] = s; red[1][tid >> 5] = s2; }
    __syncthreads();
    float ts = 0.f, ts2 = 0.f;
#pragma unroll
    for (int w = 0; w < 8; ++w) { ts += red[0][w]; ts2 += red[1][w]; }
    float mu  = ts * (1.0f / DMODEL);
    float var = ts2 * (1.0f / DMODEL) - mu * mu;
    float inv = rsqrtf(var + 1e-5f);
#pragma unroll
    for (int q = 0; q < 4; ++q) {
        int c = tid + q * 256;
        out[(size_t)row * DMODEL + c] = (yv[q] - mu) * inv * gamma[c] + beta[c];
    }
}

// ---------------------------------------------------------------------------
extern "C" void kernel_launch(void* const* d_in, const int* in_sizes, int n_in,
                              void* d_out, int out_size, void* d_ws, size_t ws_size,
                              hipStream_t stream) {
    const float* x       = (const float*)d_in[0];   // [S,B,D]
    const float* pos_emb = (const float*)d_in[1];   // [T,D]
    const float* pbu     = (const float*)d_in[2];   // [H,DH]
    const float* pbv     = (const float*)d_in[3];   // [H,DH]
    const float* memory  = (const float*)d_in[4];   // [M,B,D]
    const float* W_qkv   = (const float*)d_in[5];   // [D,3*H*DH]
    const float* W_rel   = (const float*)d_in[6];   // [D,H*DH]
    const float* W_o     = (const float*)d_in[7];   // [H*DH,D]
    const float* gamma   = (const float*)d_in[8];
    const float* beta    = (const float*)d_in[9];

    char* ws = (char*)d_ws;
    float* c_f32    = (float*)(ws);                          // 8192x1024 f32  (32 MB)
    bf16*  qkv_bf   = (bf16*)(ws + 33554432);                // 8192x3072 bf16 (48 MB)
    bf16*  r_bf     = (bf16*)(ws + 83886080);                // 2048x1024 bf16 (4 MB)
    float* attn_vec = (float*)(ws + 88080384);               // 4096x1024 f32  (16 MB)
    float* attn_out = (float*)(ws + 104857600);              // 4096x1024 f32  (16 MB)

    // c = concat(memory, x) along axis 0 -> [T*B, D]
    hipMemcpyAsync(c_f32, memory, (size_t)M_LEN * BATCH * DMODEL * 4,
                   hipMemcpyDeviceToDevice, stream);
    hipMemcpyAsync(c_f32 + (size_t)M_LEN * BATCH * DMODEL, x,
                   (size_t)S_LEN * BATCH * DMODEL * 4, hipMemcpyDeviceToDevice, stream);

    // qkv = c @ W_qkv   (8192 x 3072, K=1024) -> bf16
    gemm_tn<true><<<dim3(64, 24), 256, 0, stream>>>(c_f32, W_qkv, qkv_bf, 8192, 3072, 1024);
    // r = pos_emb @ W_rel (2048 x 1024, K=1024) -> bf16
    gemm_tn<true><<<dim3(16, 8), 256, 0, stream>>>(pos_emb, W_rel, r_bf, 2048, 1024, 1024);
    // fused relative attention
    attn_xl_kernel<<<dim3(S_LEN / 64, BATCH, HEADS), 128, 0, stream>>>(
        qkv_bf, r_bf, pbu, pbv, attn_vec);
    // attn_out = attn_vec @ W_o (4096 x 1024, K=1024) -> f32
    gemm_tn<false><<<dim3(32, 8), 256, 0, stream>>>(attn_vec, W_o, attn_out, 4096, 1024, 1024);
    // residual + layernorm
    ln_kernel<<<S_LEN * BATCH, 256, 0, stream>>>(x, attn_out, gamma, beta, (float*)d_out);
}
